// GatedAttentionFusion_13288628814169
// MI455X (gfx1250) — matrix operand order repaired and checked by reading, not compiled, not run on hardware
//
#include <hip/hip_runtime.h>

// ---------------------------------------------------------------------------
// GatedAttentionFusion for MI455X (gfx1250), bf16 WMMA pipeline.
// B=4, MAX_N=2048, D=512, H=8, HD=64 (hardwired to the reference config).
// Round 4: attention LDS tiles padded to kill bank conflicts (K rows 72,
// V/P rows 40 elems -> bank strides 36/20, conflict-free across 16 lanes);
// fragment loads batched ahead of WMMA chains for partial s_wait_dscnt
// overlap. Async double-buffered K/V staging retained.
// ---------------------------------------------------------------------------

typedef __bf16 bf16_t;
typedef bf16_t v16bf __attribute__((ext_vector_type(16)));
typedef bf16_t v8bf  __attribute__((ext_vector_type(8)));
typedef float  v8f   __attribute__((ext_vector_type(8)));

#define NB    4
#define MAXN  2048
#define DIM   512
#define NH    8
#define HD    64
#define NEGINF (-1.0e9f)
#define LOG2E 1.44269504088896340736f
#define KPAD  72   // 144B rows -> bank stride 36 (conflict-free)
#define VPAD  40   // 80B rows  -> bank stride 20 (conflict-free)

static __device__ __forceinline__ v16bf pack16(v8bf lo, v8bf hi) {
  v16bf r;
#pragma unroll
  for (int e = 0; e < 8; ++e) { r[e] = lo[e]; r[e + 8] = hi[e]; }
  return r;
}

static __device__ __forceinline__ unsigned lds_off(const void* p) {
  // Generic pointers to LDS are {shared aperture, offset}; low 32 bits = offset.
  return (unsigned)(unsigned long long)p;
}

static __device__ __forceinline__ void async_b128(unsigned lds_addr,
                                                  const void* gaddr) {
  asm volatile("global_load_async_to_lds_b128 %0, %1, off"
               :: "v"(lds_addr), "v"((unsigned long long)gaddr)
               : "memory");
}

// --------------------------- meta: counts / starts --------------------------
// meta ints: [0..3] cnt_src, [4..7] start_src, [8..11] cnt_ref, [12..15] start_ref
__global__ void meta_kernel(const int* __restrict__ bsrc, int nsrc,
                            const int* __restrict__ bref, int nref,
                            int* __restrict__ meta) {
  __shared__ int cs[NB], cr[NB];
  int tid = threadIdx.x;
  if (tid < NB) { cs[tid] = 0; cr[tid] = 0; }
  __syncthreads();
  for (int i = tid; i < nsrc; i += blockDim.x) atomicAdd(&cs[bsrc[i]], 1);
  for (int i = tid; i < nref; i += blockDim.x) atomicAdd(&cr[bref[i]], 1);
  __syncthreads();
  if (tid == 0) {
    int s = 0;
    for (int b = 0; b < NB; ++b) { meta[b] = cs[b]; meta[4 + b] = s; s += cs[b]; }
    s = 0;
    for (int b = 0; b < NB; ++b) { meta[8 + b] = cr[b]; meta[12 + b] = s; s += cr[b]; }
  }
}

// ------------------- ragged f32 -> dense zero-padded bf16 -------------------
__global__ __launch_bounds__(256) void scatter_kernel(
    const float* __restrict__ feats, const int* __restrict__ cnts,
    const int* __restrict__ starts, bf16_t* __restrict__ dense) {
  int r = blockIdx.x;                 // dense row in [0, NB*MAXN)
  int b = r >> 11, p = r & (MAXN - 1);
  int cnt = cnts[b], st = starts[b];
  const float* srow = feats + (size_t)(st + p) * DIM;
  bf16_t* drow = dense + (size_t)r * DIM;
  for (int d = threadIdx.x; d < DIM; d += blockDim.x)
    drow[d] = (p < cnt) ? (bf16_t)srow[d] : (bf16_t)0.0f;
}

__global__ void f32_to_bf16_kernel(const float* __restrict__ in,
                                   bf16_t* __restrict__ out, int n) {
  int i = blockIdx.x * blockDim.x + threadIdx.x;
  if (i < n) out[i] = (bf16_t)in[i];
}

// ------------------------------- WMMA GEMM ---------------------------------
// C[M,512] = (A[M,512](bf16) @ Wb[512,512](bf16, row-major = B^T) + bias) * scale
// MODE 0: bf16 row-major | 1: +ReLU bf16 | 2: transposed bf16 [B,H,64,2048] | 3: f32 + bf16
template <int MODE>
__global__ __launch_bounds__(256) void gemm_wmma_kernel(
    const bf16_t* __restrict__ A, const bf16_t* __restrict__ Wb,
    const float* __restrict__ bias, float scale,
    float* __restrict__ outF, bf16_t* __restrict__ outB, bf16_t* __restrict__ outT) {
  const int lane = threadIdx.x & 31;
  const int wave = threadIdx.x >> 5;
  const int half = lane >> 4;
  const int l16  = lane & 15;
  const int rowbase = blockIdx.x * 128 + wave * 16;   // 8 waves x 16 rows
  const int colbase = blockIdx.y * 64;

  v8f acc[4] = {};
#pragma unroll 2
  for (int k0 = 0; k0 < DIM; k0 += 32) {
    // A fragment (16x32): lane<16 -> K {0..7,16..23}, lane>=16 -> K {8..15,24..31}
    const bf16_t* ap = A + (size_t)(rowbase + l16) * DIM + k0 + half * 8;
    v16bf af = pack16(*(const v8bf*)ap, *(const v8bf*)(ap + 16));
#pragma unroll
    for (int g = 0; g < 4; ++g) {
      // B fragment (32x16): lane = col, K {0..7,16..23} / {8..15,24..31} like A
      const bf16_t* bp = Wb + (size_t)(colbase + g * 16 + l16) * DIM + k0 + half * 8;
      v16bf bfr = pack16(*(const v8bf*)bp, *(const v8bf*)(bp + 16));
      acc[g] = __builtin_amdgcn_wmma_f32_16x16x32_bf16(false, af, false, bfr,
                                                       (short)0, acc[g], false, false);
    }
  }
#pragma unroll
  for (int g = 0; g < 4; ++g) {
    const int col = colbase + g * 16 + l16;
    const float bv = bias[col];
    if (MODE == 2) {
      v8bf pk;
#pragma unroll
      for (int j = 0; j < 8; ++j) pk[j] = (bf16_t)((acc[g][j] + bv) * scale);
      const int h = colbase >> 6, d = col & 63;
      const int bidx = rowbase >> 11;
      const int t0 = (rowbase & (MAXN - 1)) + 8 * half;  // 8 consecutive tokens
      *(v8bf*)(outT + ((size_t)(bidx * NH + h) * HD + d) * MAXN + t0) = pk;
    } else {
#pragma unroll
      for (int j = 0; j < 8; ++j) {
        const int row = rowbase + 8 * half + j;
        float v = (acc[g][j] + bv) * scale;
        if (MODE == 1) v = fmaxf(v, 0.0f);
        if (MODE == 3) outF[(size_t)row * DIM + col] = v;
        outB[(size_t)row * DIM + col] = (bf16_t)v;
      }
    }
  }
}

// --------------------------- flash attention -------------------------------
// Q [B*2048,512] bf16 (pre-scaled by log2e/sqrt(HD) -> exp2-domain softmax),
// K [B*2048,512] bf16, Vt [B,H,64,2048] bf16.  ctx [B*2048,512] bf16.
// Block = 4 waves sharing (b,h); K/V tiles double-buffered in LDS via
// async copies; each wave owns one 16-row Q tile.
__global__ __launch_bounds__(128) void attn_kernel(
    const bf16_t* __restrict__ Q, const bf16_t* __restrict__ K,
    const bf16_t* __restrict__ Vt, const int* __restrict__ cntRef,
    bf16_t* __restrict__ ctx) {
  __shared__ alignas(32) bf16_t ldsK[2][32][KPAD];  // [buf][key][d], padded rows
  __shared__ alignas(32) bf16_t ldsV[2][64][VPAD];  // [buf][d][key], padded rows
  __shared__ alignas(32) bf16_t ldsP[4][16][VPAD];  // per-wave P bounce, padded
  const int tid  = threadIdx.x;
  const int lane = tid & 31;
  const int wave = tid >> 5;
  const int half = lane >> 4;
  const int l16  = lane & 15;
  const int b = blockIdx.x >> 3, h = blockIdx.x & 7;
  const int q0 = (blockIdx.y * 4 + wave) * 16;
  const int cnt = cntRef[b];

  const bf16_t* kbase = K + (size_t)(b * MAXN) * DIM + h * HD;
  const bf16_t* vbase = Vt + (size_t)(b * NH + h) * HD * MAXN;

  // cooperative async copy of one 32-key tile (K: 32x64, V: 64x32) into buf
  auto issue_tile = [&](int buf, int kb) {
#pragma unroll
    for (int rep = 0; rep < 2; ++rep) {
      const int s = tid + rep * 128;           // 256 x 16B segments
      const int row = s >> 3, o = (s & 7) * 8; // K row = key, 8 bf16 per seg
      async_b128(lds_off(&ldsK[buf][row][o]),
                 kbase + (size_t)(kb + row) * DIM + o);
    }
#pragma unroll
    for (int rep = 0; rep < 2; ++rep) {
      const int s = tid + rep * 128;
      const int d = s >> 2, o = (s & 3) * 8;   // V row = feature d
      async_b128(lds_off(&ldsV[buf][d][o]),
                 vbase + (size_t)d * MAXN + kb + o);
    }
  };

  // Q A-fragments over head dim (two K=32 slabs)
  const bf16_t* qp = Q + (size_t)(b * MAXN + q0 + l16) * DIM + h * HD;
  v16bf qa[2];
#pragma unroll
  for (int wi = 0; wi < 2; ++wi) {
    const bf16_t* p = qp + wi * 32 + half * 8;
    qa[wi] = pack16(*(const v8bf*)p, *(const v8bf*)(p + 16));
  }

  // all-ones B fragment: P @ ones accumulates row sums into C layout
  v16bf ones;
#pragma unroll
  for (int e = 0; e < 16; ++e) ones[e] = (bf16_t)1.0f;

  float row_m[8];
#pragma unroll
  for (int j = 0; j < 8; ++j) row_m[j] = -1.0e30f;
  v8f lacc = {};          // running row sums (exp2 domain), lane-replicated
  v8f o[4] = {};

  const int ntiles = (cnt + 31) >> 5;
  issue_tile(0, 0);                    // prologue: tile 0 in flight

  for (int kt = 0; kt < ntiles; ++kt) {
    const int kb = kt * 32;
    const int buf = kt & 1;
    asm volatile("s_wait_asynccnt 0x0" ::: "memory");  // our tile-kt copies done
    __syncthreads();  // tile kt visible block-wide; buffer (kt+1)&1 is free
    if (kt + 1 < ntiles) issue_tile(buf ^ 1, kb + 32); // overlap with consume
    // ---- batch-load all K fragments, then run the score WMMA chain ----
    v16bf kf[2][2];
#pragma unroll
    for (int g = 0; g < 2; ++g)
#pragma unroll
      for (int wi = 0; wi < 2; ++wi) {
        const bf16_t* p = &ldsK[buf][g * 16 + l16][wi * 32 + half * 8];
        kf[g][wi] = pack16(*(const v8bf*)p, *(const v8bf*)(p + 16));
      }
    v8f s[2] = {};
#pragma unroll
    for (int g = 0; g < 2; ++g)
#pragma unroll
      for (int wi = 0; wi < 2; ++wi)
        s[g] = __builtin_amdgcn_wmma_f32_16x16x32_bf16(false, qa[wi], false,
                                                       kf[g][wi], (short)0,
                                                       s[g], false, false);
    const bool m0 = (kb + l16) >= cnt;
    const bool m1 = (kb + 16 + l16) >= cnt;
    // ---- online softmax stats (exp2 domain) ----
#pragma unroll
    for (int j = 0; j < 8; ++j) {
      const float s0 = m0 ? NEGINF : s[0][j];
      const float s1 = m1 ? NEGINF : s[1][j];
      float t = fmaxf(s0, s1);
#pragma unroll
      for (int ofs = 1; ofs < 16; ofs <<= 1) t = fmaxf(t, __shfl_xor(t, ofs, 32));
      const float mn = fmaxf(row_m[j], t);
      const float corr = __builtin_amdgcn_exp2f(row_m[j] - mn);
      row_m[j] = mn;
      lacc[j] *= corr;
#pragma unroll
      for (int g2 = 0; g2 < 4; ++g2) o[g2][j] *= corr;
      ldsP[wave][j + 8 * half][l16]      = (bf16_t)__builtin_amdgcn_exp2f(s0 - mn);
      ldsP[wave][j + 8 * half][16 + l16] = (bf16_t)__builtin_amdgcn_exp2f(s1 - mn);
    }
    // ---- re-read P in A-fragment layout; batch V fragments ----
    v16bf pf = pack16(*(const v8bf*)&ldsP[wave][l16][half * 8],
                      *(const v8bf*)&ldsP[wave][l16][16 + half * 8]);
    v16bf vf[4];
#pragma unroll
    for (int g2 = 0; g2 < 4; ++g2) {
      const bf16_t* p = &ldsV[buf][g2 * 16 + l16][half * 8];
      vf[g2] = pack16(*(const v8bf*)p, *(const v8bf*)(p + 16));
    }
    // row-sum via WMMA: lacc += P @ ones  (replaces shuffle-tree reduction)
    lacc = __builtin_amdgcn_wmma_f32_16x16x32_bf16(false, pf, false, ones,
                                                   (short)0, lacc, false, false);
    // ---- O += P V ----
#pragma unroll
    for (int g2 = 0; g2 < 4; ++g2)
      o[g2] = __builtin_amdgcn_wmma_f32_16x16x32_bf16(false, pf, false, vf[g2],
                                                      (short)0, o[g2], false, false);
  }
  // ---- normalize + store ctx ----
  float inv[8];
#pragma unroll
  for (int j = 0; j < 8; ++j) inv[j] = __builtin_amdgcn_rcpf(lacc[j]);
#pragma unroll
  for (int g2 = 0; g2 < 4; ++g2)
#pragma unroll
    for (int j = 0; j < 8; ++j) {
      const int row = b * MAXN + q0 + 8 * half + j;
      const int col = h * HD + g2 * 16 + l16;
      ctx[(size_t)row * DIM + col] = (bf16_t)(o[g2][j] * inv[j]);
    }
}

// --------------------------- gate matvec + sigmoid --------------------------
__global__ __launch_bounds__(256) void gate_kernel(
    const bf16_t* __restrict__ h1, const float* __restrict__ gw2,
    const float* __restrict__ gb2, float* __restrict__ gate) {
  const int lane = threadIdx.x & 31;
  const int row = blockIdx.x * 8 + (threadIdx.x >> 5);
  const bf16_t* hp = h1 + (size_t)row * DIM;
  float s = 0.0f;
#pragma unroll
  for (int i = 0; i < DIM / 32; ++i) s += (float)hp[lane + i * 32] * gw2[lane + i * 32];
#pragma unroll
  for (int ofs = 1; ofs < 32; ofs <<= 1) s += __shfl_xor(s, ofs, 32);
  if (lane == 0) {
    const float z = s + gb2[0];
    gate[row] = __builtin_amdgcn_rcpf(1.0f + __builtin_amdgcn_exp2f(-z * LOG2E));
  }
}

// ------------------------ residual + ragged gather --------------------------
__global__ __launch_bounds__(256) void out_kernel(
    const float* __restrict__ feats_src, const int* __restrict__ bsrc,
    const int* __restrict__ startsSrc, const float* __restrict__ alignedF,
    const float* __restrict__ gate, float* __restrict__ out_flat, int nsrc) {
  const int idx = blockIdx.x * blockDim.x + threadIdx.x;
  if (idx >= nsrc * DIM) return;
  const int i = idx >> 9, d = idx & (DIM - 1);
  const int b = bsrc[i];
  const int r = b * MAXN + (i - startsSrc[b]);
  out_flat[idx] = feats_src[idx] + gate[r] * alignedF[(size_t)r * DIM + d];
}

__global__ void gate_out_kernel(const float* __restrict__ gate,
                                float* __restrict__ outg) {
  const int i = blockIdx.x * blockDim.x + threadIdx.x;
  if (i < NB * MAXN) outg[i] = gate[i];
}

// ------------------------------- launcher ----------------------------------
extern "C" void kernel_launch(void* const* d_in, const int* in_sizes, int n_in,
                              void* d_out, int out_size, void* d_ws, size_t ws_size,
                              hipStream_t stream) {
  const float* feats_src = (const float*)d_in[0];
  const int*   batch_src = (const int*)d_in[1];
  const float* feats_ref = (const float*)d_in[2];
  const int*   batch_ref = (const int*)d_in[3];
  const float* w_in  = (const float*)d_in[4];
  const float* b_in  = (const float*)d_in[5];
  const float* w_out = (const float*)d_in[6];
  const float* b_out = (const float*)d_in[7];
  const float* gw1   = (const float*)d_in[8];
  const float* gb1   = (const float*)d_in[9];
  const float* gw2   = (const float*)d_in[10];
  const float* gb2   = (const float*)d_in[11];
  const int n_src = in_sizes[1];
  const int n_ref = in_sizes[3];

  const int ROWS = NB * MAXN;                 // 8192
  char* ws = (char*)d_ws;
  size_t off = 0;
  auto alloc = [&](size_t bytes) { char* p = ws + off; off += (bytes + 255) & ~(size_t)255; return p; };

  int*    meta   = (int*)   alloc(64 * sizeof(int));
  bf16_t* xsrc   = (bf16_t*)alloc((size_t)ROWS * DIM * 2);
  bf16_t* xref   = (bf16_t*)alloc((size_t)ROWS * DIM * 2);
  bf16_t* qb     = (bf16_t*)alloc((size_t)ROWS * DIM * 2);
  bf16_t* kb     = (bf16_t*)alloc((size_t)ROWS * DIM * 2);
  bf16_t* vtb    = (bf16_t*)alloc((size_t)ROWS * DIM * 2);   // [B,H,64,2048]
  bf16_t* ctx    = (bf16_t*)alloc((size_t)ROWS * DIM * 2);
  float*  alignF = (float*) alloc((size_t)ROWS * DIM * 4);
  bf16_t* alignB = (bf16_t*)alloc((size_t)ROWS * DIM * 2);
  bf16_t* h1     = (bf16_t*)alloc((size_t)ROWS * DIM * 2);
  float*  gate   = (float*) alloc((size_t)ROWS * 4);
  bf16_t* w_in_b  = (bf16_t*)alloc((size_t)3 * DIM * DIM * 2);
  bf16_t* w_out_b = (bf16_t*)alloc((size_t)DIM * DIM * 2);
  bf16_t* gw1_b   = (bf16_t*)alloc((size_t)DIM * DIM * 2);
  (void)ws_size; (void)n_in;

  meta_kernel<<<1, 256, 0, stream>>>(batch_src, n_src, batch_ref, n_ref, meta);
  scatter_kernel<<<ROWS, 256, 0, stream>>>(feats_src, meta + 0, meta + 4, xsrc);
  scatter_kernel<<<ROWS, 256, 0, stream>>>(feats_ref, meta + 8, meta + 12, xref);
  f32_to_bf16_kernel<<<(3 * DIM * DIM + 255) / 256, 256, 0, stream>>>(w_in, w_in_b, 3 * DIM * DIM);
  f32_to_bf16_kernel<<<(DIM * DIM + 255) / 256, 256, 0, stream>>>(w_out, w_out_b, DIM * DIM);
  f32_to_bf16_kernel<<<(DIM * DIM + 255) / 256, 256, 0, stream>>>(gw1, gw1_b, DIM * DIM);

  dim3 ggrid(ROWS / 128, DIM / 64);
  // Q scale folds 1/sqrt(HD) AND log2(e) (exp2-domain softmax); K row-major; V transposed
  gemm_wmma_kernel<0><<<ggrid, 256, 0, stream>>>(xsrc, w_in_b, b_in, 0.125f * LOG2E,
                                                 nullptr, qb, nullptr);
  gemm_wmma_kernel<0><<<ggrid, 256, 0, stream>>>(xref, w_in_b + (size_t)DIM * DIM,
                                                 b_in + DIM, 1.0f, nullptr, kb, nullptr);
  gemm_wmma_kernel<2><<<ggrid, 256, 0, stream>>>(xref, w_in_b + (size_t)2 * DIM * DIM,
                                                 b_in + 2 * DIM, 1.0f, nullptr, nullptr, vtb);

  attn_kernel<<<dim3(NB * NH, MAXN / 64), 128, 0, stream>>>(qb, kb, vtb, meta + 8, ctx);

  gemm_wmma_kernel<3><<<ggrid, 256, 0, stream>>>(ctx, w_out_b, b_out, 1.0f,
                                                 alignF, alignB, nullptr);
  gemm_wmma_kernel<1><<<ggrid, 256, 0, stream>>>(alignB, gw1_b, gb1, 1.0f,
                                                 nullptr, h1, nullptr);

  gate_kernel<<<ROWS / 8, 256, 0, stream>>>(h1, gw2, gb2, gate);

  float* out_flat = (float*)d_out;
  float* out_gate = out_flat + (size_t)n_src * DIM;
  out_kernel<<<(n_src * DIM + 255) / 256, 256, 0, stream>>>(
      feats_src, batch_src, meta + 4, alignF, gate, out_flat, n_src);
  gate_out_kernel<<<ROWS / 256, 256, 0, stream>>>(gate, out_gate);
}
